// Contrast_88012469829980
// MI455X (gfx1250) — matrix operand order repaired and checked by reading, not compiled, not run on hardware
//
#include <hip/hip_runtime.h>
#include <hip/hip_bf16.h>

// ---------------------------------------------------------------------------
// Problem constants (match reference)
// ---------------------------------------------------------------------------
#define NROW 4096
#define DIM  1024
#define GCL_INV_TAU 1.25f       // 1/0.8
#define NCE_INV_TAU 10.0f       // 1/0.1
#define NUM_NEG 100

typedef __attribute__((ext_vector_type(16))) __bf16 v16bf;
typedef __attribute__((ext_vector_type(8)))  __bf16 v8bf;
typedef __attribute__((ext_vector_type(8)))  float  v8f;

union AFrag { v16bf v; v8bf h[2]; };

__device__ __forceinline__ v8f v8f_zero() {
    v8f a; for (int e = 0; e < 8; ++e) a[e] = 0.f; return a;
}

// A fragment: 16x32 bf16, row = m0 + (lane&15).
// lanes 0-15 : elems[0..7]=K k0+0..7,  elems[8..15]=K k0+16..23
// lanes 16-31: elems[0..7]=K k0+8..15, elems[8..15]=K k0+24..31
__device__ __forceinline__ v16bf load_a_frag(const __hip_bfloat16* A, int row,
                                             int k0, int ldk, int half) {
    const __hip_bfloat16* p = A + (size_t)row * ldk + k0 + half * 8;
    AFrag a;
    a.h[0] = *(const v8bf*)(const void*)(p);
    a.h[1] = *(const v8bf*)(const void*)(p + 16);
    return a.v;
}

// B fragment: 32x16 bf16 (KxN), col row of Bt = brow. Same K order as A:
// lanes 0-15: K k0+0..7, k0+16..23; lanes 16-31: K k0+8..15, k0+24..31.
__device__ __forceinline__ v16bf load_b_frag(const __hip_bfloat16* B, int brow,
                                             int k0, int ldk, int half) {
    const __hip_bfloat16* p = B + (size_t)brow * ldk + k0 + half * 8;
    AFrag b;
    b.h[0] = *(const v8bf*)(const void*)(p);
    b.h[1] = *(const v8bf*)(const void*)(p + 16);
    return b.v;
}

// ---------------------------------------------------------------------------
// Utility kernels
// ---------------------------------------------------------------------------
__global__ void k_zero(float* p, int n) {
    int i = blockIdx.x * blockDim.x + threadIdx.x;
    if (i < n) p[i] = 0.f;
}

__global__ void k_f32_to_bf16(const float* __restrict__ x,
                              __hip_bfloat16* __restrict__ y, int n) {
    int i = blockIdx.x * blockDim.x + threadIdx.x;
    if (i < n) y[i] = __float2bfloat16(x[i]);
}

__global__ void k_blend_bf16(const float* __restrict__ a, const float* __restrict__ b,
                             const float* __restrict__ gptr,
                             __hip_bfloat16* __restrict__ y, int n) {
    int i = blockIdx.x * blockDim.x + threadIdx.x;
    if (i < n) {
        float g = *gptr;
        y[i] = __float2bfloat16((1.f - g) * a[i] + g * b[i]);
    }
}

// ---------------------------------------------------------------------------
// GEMM: C[M,Nc] = act(A[M,K](bf16) * B[Nc,K](bf16)^T + bias)
// block = 256 threads (8 waves); wave -> 16 rows x 64 cols; grid (M/128, Nc/64)
// elu!=0 -> ELU + store bf16 into outB; else raw f32 into outF.
// ---------------------------------------------------------------------------
__global__ void k_gemm_bt(const __hip_bfloat16* __restrict__ A,
                          const __hip_bfloat16* __restrict__ B,
                          const float* __restrict__ bias,
                          float* __restrict__ outF,
                          __hip_bfloat16* __restrict__ outB,
                          int Nc, int K, int elu) {
    const int lane = threadIdx.x & 31;
    const int wave = threadIdx.x >> 5;
    const int half = lane >> 4;
    const int lr   = lane & 15;
    const int m0 = blockIdx.x * 128 + wave * 16;
    const int n0 = blockIdx.y * 64;

    v8f acc[4];
    for (int t = 0; t < 4; ++t) acc[t] = v8f_zero();

    const int arow = m0 + lr;
    for (int k0 = 0; k0 < K; k0 += 32) {
        v16bf a = load_a_frag(A, arow, k0, K, half);
        for (int t = 0; t < 4; ++t) {
            v16bf b = load_b_frag(B, n0 + t * 16 + lr, k0, K, half);
            acc[t] = __builtin_amdgcn_wmma_f32_16x16x32_bf16(
                false, a, false, b, (short)0, acc[t], false, false);
        }
    }
    for (int t = 0; t < 4; ++t) {
        int col = n0 + t * 16 + lr;
        float bv = bias[col];
        for (int r = 0; r < 8; ++r) {
            int row = m0 + half * 8 + r;
            float v = acc[t][r] + bv;
            if (elu) {
                v = (v > 0.f) ? v : expm1f(v);
                outB[(size_t)row * Nc + col] = __float2bfloat16(v);
            } else {
                outF[(size_t)row * Nc + col] = v;
            }
        }
    }
}

// Row-normalize f32 matrix -> bf16 (one wave per row).
__global__ void k_rownorm(const float* __restrict__ P,
                          __hip_bfloat16* __restrict__ Pn) {
    const int lane = threadIdx.x & 31;
    const int row  = blockIdx.x * (blockDim.x >> 5) + (threadIdx.x >> 5);
    const float* pr = P + (size_t)row * DIM;
    float s = 0.f;
    for (int k = lane; k < DIM; k += 32) { float v = pr[k]; s += v * v; }
    for (int m = 16; m >= 1; m >>= 1) s += __shfl_xor(s, m, 32);
    float inv = rsqrtf(s + 1e-30f);
    __hip_bfloat16* out = Pn + (size_t)row * DIM;
    for (int k = lane; k < DIM; k += 32) out[k] = __float2bfloat16(pr[k] * inv);
}

// ---------------------------------------------------------------------------
// Fused similarity + reductions for one GCL loss.
// m[i,j] = exp(cos(i,j)/tau); accumulate:
//   rowsum[i] = sum_j m, rowpos[i] = sum_j m*pos[i,j]          (registers)
//   colsum[j] = sum_i m, colpos[j] = sum_i m*pos[j,i]          (atomics)
// grid = 32 blocks x 256 threads; wave owns 16 rows, sweeps all 4096 cols.
// ---------------------------------------------------------------------------
__global__ void k_sim_reduce(const __hip_bfloat16* __restrict__ Pm,
                             const __hip_bfloat16* __restrict__ Ps,
                             const float* __restrict__ pos,
                             float* __restrict__ rowsum, float* __restrict__ rowpos,
                             float* __restrict__ colsum, float* __restrict__ colpos) {
    const int lane = threadIdx.x & 31;
    const int wave = threadIdx.x >> 5;
    const int half = lane >> 4;
    const int lr   = lane & 15;
    const int r0 = blockIdx.x * 128 + wave * 16;
    const int arow = r0 + lr;

    float rs[8], rp[8];
    for (int r = 0; r < 8; ++r) { rs[r] = 0.f; rp[r] = 0.f; }

    for (int c0 = 0; c0 < NROW; c0 += 16) {
        v8f acc = v8f_zero();
        for (int k0 = 0; k0 < DIM; k0 += 32) {
            v16bf a = load_a_frag(Pm, arow, k0, DIM, half);
            v16bf b = load_b_frag(Ps, c0 + lr, k0, DIM, half);
            acc = __builtin_amdgcn_wmma_f32_16x16x32_bf16(
                false, a, false, b, (short)0, acc, false, false);
        }
        float cs = 0.f, cp = 0.f;
        const int col = c0 + lr;
        for (int r = 0; r < 8; ++r) {
            int row = r0 + half * 8 + r;
            float m = expf(acc[r] * GCL_INV_TAU);
            float pv  = pos[(size_t)row * NROW + col];   // pos[row,col]
            float pvt = pos[(size_t)col * NROW + row];   // pos[col,row]
            rs[r] += m; rp[r] += m * pv;
            cs += m;    cp += m * pvt;
        }
        cs += __shfl_xor(cs, 16, 32);
        cp += __shfl_xor(cp, 16, 32);
        if (half == 0) {
            atomicAdd(&colsum[col], cs);
            atomicAdd(&colpos[col], cp);
        }
    }
    for (int r = 0; r < 8; ++r) {
        float a = rs[r], b = rp[r];
        for (int m = 8; m >= 1; m >>= 1) {
            a += __shfl_xor(a, m, 32);
            b += __shfl_xor(b, m, 32);
        }
        if (lr == 0) {
            int row = r0 + half * 8 + r;
            rowsum[row] = a;
            rowpos[row] = b;
        }
    }
}

__global__ void k_gcl_finalize(const float* __restrict__ rs, const float* __restrict__ rp,
                               const float* __restrict__ cs, const float* __restrict__ cp,
                               float* __restrict__ acc) {
    __shared__ float red[256];
    float s = 0.f;
    for (int i = threadIdx.x; i < NROW; i += 256) {
        s += 0.5f * (-logf(rp[i] / (rs[i] + 1e-8f)));   // LAM * lori_mp term
        s += 0.5f * (-logf(cp[i] / (cs[i] + 1e-8f)));   // (1-LAM) * lori_sc term
    }
    red[threadIdx.x] = s;
    __syncthreads();
    for (int st = 128; st > 0; st >>= 1) {
        if (threadIdx.x < st) red[threadIdx.x] += red[threadIdx.x + st];
        __syncthreads();
    }
    if (threadIdx.x == 0) atomicAdd(&acc[0], red[0] / (float)NROW);
}

// ---------------------------------------------------------------------------
// InfoNCE: per-row logsumexp over NUM_NEG sampled negative columns (md==0).
// One wave per row; deterministic counter hash + linear-probe rejection.
// ---------------------------------------------------------------------------
__global__ void k_neg_lse(const __hip_bfloat16* __restrict__ z1,
                          const __hip_bfloat16* __restrict__ z2,
                          const float* __restrict__ md,
                          float* __restrict__ lse) {
    const int lane = threadIdx.x & 31;
    const int i = blockIdx.x * (blockDim.x >> 5) + (threadIdx.x >> 5);
    const __hip_bfloat16* zr = z1 + (size_t)i * DIM;
    float mx = -1e30f, sm = 0.f;
    for (int t = 0; t < NUM_NEG; ++t) {
        unsigned h = (unsigned)i * 2654435761u + (unsigned)t * 2246822519u + 0x9e3779b9u;
        h ^= h >> 16; h *= 2246822519u; h ^= h >> 13; h *= 3266489917u; h ^= h >> 16;
        int j = (int)(h & (NROW - 1));
        while (md[(size_t)i * NROW + j] > 0.f) j = (j + 1) & (NROW - 1);
        const __hip_bfloat16* zc = z2 + (size_t)j * DIM;
        float d = 0.f;
        for (int k = lane; k < DIM; k += 32)
            d += __bfloat162float(zr[k]) * __bfloat162float(zc[k]);
        for (int m = 16; m >= 1; m >>= 1) d += __shfl_xor(d, m, 32);
        float p = d * NCE_INV_TAU;
        if (p > mx) { sm = sm * expf(mx - p) + 1.f; mx = p; }
        else         { sm += expf(p - mx); }
    }
    if (lane == 0) lse[i] = mx + logf(sm);
}

// ---------------------------------------------------------------------------
// InfoNCE positive pass: s = z1*z2^T tiles (WMMA), accumulate
//   num = sum md * (logaddexp(p, lse_neg[row]) - p),  den = sum md
// ---------------------------------------------------------------------------
__global__ void k_pos_pair(const __hip_bfloat16* __restrict__ z1,
                           const __hip_bfloat16* __restrict__ z2,
                           const float* __restrict__ md,
                           const float* __restrict__ lse,
                           float* __restrict__ acc) {
    const int lane = threadIdx.x & 31;
    const int wave = threadIdx.x >> 5;
    const int half = lane >> 4;
    const int lr   = lane & 15;
    const int r0 = blockIdx.x * 128 + wave * 16;
    const int arow = r0 + lr;

    float Lr[8];
    for (int r = 0; r < 8; ++r) Lr[r] = lse[r0 + half * 8 + r];

    float num = 0.f, den = 0.f;
    for (int c0 = 0; c0 < NROW; c0 += 16) {
        v8f acc8 = v8f_zero();
        for (int k0 = 0; k0 < DIM; k0 += 32) {
            v16bf a = load_a_frag(z1, arow, k0, DIM, half);
            v16bf b = load_b_frag(z2, c0 + lr, k0, DIM, half);
            acc8 = __builtin_amdgcn_wmma_f32_16x16x32_bf16(
                false, a, false, b, (short)0, acc8, false, false);
        }
        const int col = c0 + lr;
        for (int r = 0; r < 8; ++r) {
            int row = r0 + half * 8 + r;
            float w = md[(size_t)row * NROW + col];
            den += w;
            if (w > 0.f) {
                float p  = acc8[r] * NCE_INV_TAU;
                float hi = fmaxf(p, Lr[r]);
                float lo = fminf(p, Lr[r]);
                float per = hi + log1pf(expf(lo - hi)) - p;
                num += per * w;
            }
        }
    }
    for (int m = 16; m >= 1; m >>= 1) {
        num += __shfl_xor(num, m, 32);
        den += __shfl_xor(den, m, 32);
    }
    if (lane == 0) {
        atomicAdd(&acc[1], num);
        atomicAdd(&acc[2], den);
    }
}

__global__ void k_combine(const float* __restrict__ acc, float* __restrict__ out) {
    out[0] = acc[0] + acc[1] / acc[2];
}

// ---------------------------------------------------------------------------
// Host launch
// ---------------------------------------------------------------------------
extern "C" void kernel_launch(void* const* d_in, const int* in_sizes, int n_in,
                              void* d_out, int out_size, void* d_ws, size_t ws_size,
                              hipStream_t stream) {
    (void)in_sizes; (void)n_in; (void)out_size; (void)ws_size;
    const float* z_mp1 = (const float*)d_in[0];
    const float* z_sc1 = (const float*)d_in[1];
    const float* pos1  = (const float*)d_in[2];
    const float* z_mp2 = (const float*)d_in[3];
    const float* z_sc2 = (const float*)d_in[4];
    const float* pos2  = (const float*)d_in[5];
    const float* md    = (const float*)d_in[6];
    const float* gamma = (const float*)d_in[7];
    const float* W1    = (const float*)d_in[8];
    const float* b1    = (const float*)d_in[9];
    const float* W2    = (const float*)d_in[10];
    const float* b2    = (const float*)d_in[11];
    float* out = (float*)d_out;

    // workspace carve-up (256B aligned)
    char* ws = (char*)d_ws;
    size_t o = 0;
    auto carve = [&](size_t bytes) -> char* {
        char* p = ws + o;
        o = (o + bytes + 255) & ~(size_t)255;
        return p;
    };
    const size_t ND2 = (size_t)NROW * DIM * 2;   // bf16 matrix bytes
    const size_t DD2 = (size_t)DIM * DIM * 2;
    __hip_bfloat16* Wb1 = (__hip_bfloat16*)carve(DD2);
    __hip_bfloat16* Wb2 = (__hip_bfloat16*)carve(DD2);
    __hip_bfloat16* Zb  = (__hip_bfloat16*)carve(ND2);
    __hip_bfloat16* Hb  = (__hip_bfloat16*)carve(ND2);
    float*          Pf  = (float*)carve((size_t)NROW * DIM * 4);   // aliased later
    __hip_bfloat16* Pn1 = (__hip_bfloat16*)carve(ND2);
    __hip_bfloat16* Pn2 = (__hip_bfloat16*)carve(ND2);
    float* rowsum = (float*)carve(NROW * 4);
    float* rowpos = (float*)carve(NROW * 4);
    float* colsum = (float*)carve(NROW * 4 * 2);   // colsum + colpos contiguous
    float* colpos = colsum + NROW;
    float* lse    = (float*)carve(NROW * 4);
    float* acc    = (float*)carve(16 * 4);

    // bf16 aliases of Pf region for the blended InfoNCE embeddings
    __hip_bfloat16* zb1 = (__hip_bfloat16*)Pf;
    __hip_bfloat16* zb2 = (__hip_bfloat16*)((char*)Pf + ND2);

    const int TB = 256;
    const int nZ = NROW * DIM;
    const int nW = DIM * DIM;
    dim3 gemmGrid(NROW / 128, DIM / 64);   // projections: M=4096, Nc=1024
    dim3 simGrid(NROW / 128);

    k_zero<<<1, 64, 0, stream>>>(acc, 16);
    k_f32_to_bf16<<<nW / TB, TB, 0, stream>>>(W1, Wb1, nW);
    k_f32_to_bf16<<<nW / TB, TB, 0, stream>>>(W2, Wb2, nW);

    for (int l = 0; l < 2; ++l) {
        const float* zmp = l ? z_mp2 : z_mp1;
        const float* zsc = l ? z_sc2 : z_sc1;
        const float* pos = l ? pos2  : pos1;

        // p_mp
        k_f32_to_bf16<<<nZ / TB, TB, 0, stream>>>(zmp, Zb, nZ);
        k_gemm_bt<<<gemmGrid, TB, 0, stream>>>(Zb, Wb1, b1, nullptr, Hb, DIM, DIM, 1);
        k_gemm_bt<<<gemmGrid, TB, 0, stream>>>(Hb, Wb2, b2, Pf, nullptr, DIM, DIM, 0);
        k_rownorm<<<NROW / 8, TB, 0, stream>>>(Pf, Pn1);
        // p_sc
        k_f32_to_bf16<<<nZ / TB, TB, 0, stream>>>(zsc, Zb, nZ);
        k_gemm_bt<<<gemmGrid, TB, 0, stream>>>(Zb, Wb1, b1, nullptr, Hb, DIM, DIM, 1);
        k_gemm_bt<<<gemmGrid, TB, 0, stream>>>(Hb, Wb2, b2, Pf, nullptr, DIM, DIM, 0);
        k_rownorm<<<NROW / 8, TB, 0, stream>>>(Pf, Pn2);

        k_zero<<<(2 * NROW) / TB, TB, 0, stream>>>(colsum, 2 * NROW);
        k_sim_reduce<<<simGrid, TB, 0, stream>>>(Pn1, Pn2, pos,
                                                 rowsum, rowpos, colsum, colpos);
        k_gcl_finalize<<<1, TB, 0, stream>>>(rowsum, rowpos, colsum, colpos, acc);
    }

    // InfoNCE (Pf region now free -> holds zb1/zb2 bf16)
    k_blend_bf16<<<nZ / TB, TB, 0, stream>>>(z_mp1, z_sc1, gamma, zb1, nZ);
    k_blend_bf16<<<nZ / TB, TB, 0, stream>>>(z_mp2, z_sc2, gamma, zb2, nZ);
    k_neg_lse<<<NROW / 8, TB, 0, stream>>>(zb1, zb2, md, lse);
    k_pos_pair<<<simGrid, TB, 0, stream>>>(zb1, zb2, md, lse, acc);
    k_combine<<<1, 1, 0, stream>>>(acc, out);
}
